// CNNVoronoi_82351702933738
// MI455X (gfx1250) — matrix operand order repaired and checked by reading, not compiled, not run on hardware
//
#include <hip/hip_runtime.h>

typedef __bf16 bf16;
typedef __attribute__((ext_vector_type(16))) __bf16 v16bf;
typedef __attribute__((ext_vector_type(8)))  float  v8f;

union Frag32B {
    uint4  u[2];
    v16bf  v;
};

#define BIGF 3.0e38f

// ---------------------------------------------------------------------------
// Prep kernel 1: w2 [64][32][3][3] fp32  ->  w2t [tap(9)][o(64)][c(32)] bf16
// ---------------------------------------------------------------------------
__global__ void prep_w2_kernel(const float* __restrict__ w2, bf16* __restrict__ w2t) {
    int i = blockIdx.x * 256 + threadIdx.x;
    if (i < 9 * 64 * 32) {
        int c   = i & 31;
        int o   = (i >> 5) & 63;
        int tap = i >> 11;
        w2t[i] = (bf16)w2[(o * 32 + c) * 9 + tap];
    }
}

// ---------------------------------------------------------------------------
// Prep kernel 2: centroids fp32 [1000][4096] -> bf16 ct [1024][4096] (zero pad)
//                + c2[n] = sum d c[n,d]^2  (pad rows -> BIGF)
// ---------------------------------------------------------------------------
__global__ void prep_cent_kernel(const float* __restrict__ cen,
                                 bf16* __restrict__ ct, float* __restrict__ c2) {
    int n = blockIdx.x;
    int t = threadIdx.x;
    __shared__ float red[256];
    float s = 0.f;
    if (n < 1000) {
        #pragma unroll
        for (int i = 0; i < 16; ++i) {
            int d = i * 256 + t;
            float v = cen[n * 4096 + d];
            s += v * v;
            ct[n * 4096 + d] = (bf16)v;
        }
    } else {
        #pragma unroll
        for (int i = 0; i < 16; ++i) {
            int d = i * 256 + t;
            ct[n * 4096 + d] = (bf16)0.f;
        }
    }
    red[t] = s;
    __syncthreads();
    for (int st = 128; st > 0; st >>= 1) {
        if (t < st) red[t] += red[t + st];
        __syncthreads();
    }
    if (t == 0) c2[n] = (n < 1000) ? red[0] : BIGF;
}

// ---------------------------------------------------------------------------
// Fused per-image kernel: conv1(fp32 VALU)+relu+pool -> LDS bf16 padded tile
//                         conv2 as implicit GEMM via v_wmma_f32_16x16x32_bf16
//                         bias+relu+pool -> features f [b][4096] bf16
// One workgroup (256 thr = 8 waves) per image.
// ---------------------------------------------------------------------------
__global__ void __launch_bounds__(256)
conv_fused_kernel(const float* __restrict__ x,    // [B][3][32][32]
                  const float* __restrict__ w1,   // [32][3][3][3]
                  const float* __restrict__ b1,   // [32]
                  const float* __restrict__ b2,   // [64]
                  const bf16*  __restrict__ w2t,  // [9][64][32]
                  bf16* __restrict__ f)           // [B][4096]
{
    // LDS: region A (32768 B) = union{ xs[3072]f32+w1s[896]f32 | prepool[16384]bf16 }
    //      region B = padded conv1 output tile [18][18][32] bf16
    __shared__ __align__(16) unsigned char smemA[32768];
    __shared__ __align__(16) bf16 h1s[18 * 18 * 32];

    float* xs      = (float*)smemA;            // 3*32*32 = 3072 floats
    float* w1s     = (float*)(smemA + 12288);  // 864 weights + 32 bias
    bf16*  prepool = (bf16*)smemA;             // 256 pos * 64 ch

    const int b = blockIdx.x;
    const int t = threadIdx.x;

    // ---- Phase 1: stage input image + conv1 weights, zero the padded tile ----
    {
        const float4* xg4 = (const float4*)(x + (size_t)b * 3072);
        float4* xs4 = (float4*)xs;
        #pragma unroll
        for (int i = t; i < 768; i += 256) xs4[i] = xg4[i];
        for (int i = t; i < 864; i += 256) w1s[i] = w1[i];
        if (t < 32) w1s[864 + t] = b1[t];
        uint4 z = make_uint4(0, 0, 0, 0);
        uint4* h4 = (uint4*)h1s;
        for (int i = t; i < 1296; i += 256) h4[i] = z;   // 18*18*32*2 / 16
    }
    __syncthreads();

    // ---- Phase 2: conv1 + relu + 2x2 pool, one (y,x) position per thread ----
    {
        const int y = t >> 4, xo = t & 15;
        float px[3][4][4];
        #pragma unroll
        for (int ic = 0; ic < 3; ++ic)
            #pragma unroll
            for (int iy = 0; iy < 4; ++iy) {
                int Y = 2 * y - 1 + iy;
                #pragma unroll
                for (int ix = 0; ix < 4; ++ix) {
                    int X = 2 * xo - 1 + ix;
                    px[ic][iy][ix] = (Y >= 0 && Y < 32 && X >= 0 && X < 32)
                                         ? xs[(ic * 32 + Y) * 32 + X] : 0.f;
                }
            }
        bf16* hdst = &h1s[((y + 1) * 18 + (xo + 1)) * 32];
        for (int oc = 0; oc < 32; ++oc) {
            float a00 = 0.f, a01 = 0.f, a10 = 0.f, a11 = 0.f;
            #pragma unroll
            for (int ic = 0; ic < 3; ++ic)
                #pragma unroll
                for (int r = 0; r < 3; ++r)
                    #pragma unroll
                    for (int s = 0; s < 3; ++s) {
                        float w = w1s[((oc * 3 + ic) * 3 + r) * 3 + s];
                        a00 = fmaf(px[ic][r][s],         w, a00);
                        a01 = fmaf(px[ic][r][s + 1],     w, a01);
                        a10 = fmaf(px[ic][r + 1][s],     w, a10);
                        a11 = fmaf(px[ic][r + 1][s + 1], w, a11);
                    }
            float bz = w1s[864 + oc];
            float m = fmaxf(fmaxf(a00, a01), fmaxf(a10, a11)) + bz;
            m = fmaxf(m, 0.f);
            hdst[oc] = (bf16)m;
        }
    }
    __syncthreads();

    // ---- Phase 3: conv2 implicit GEMM, K = 9 taps x 32 channels ----
    const int wv = t >> 5, lane = t & 31;
    const int lane15 = lane & 15, grp = lane >> 4;

    v8f acc[2][4];
    v8f vzero = {0.f, 0.f, 0.f, 0.f, 0.f, 0.f, 0.f, 0.f};
    #pragma unroll
    for (int mi = 0; mi < 2; ++mi)
        #pragma unroll
        for (int nt = 0; nt < 4; ++nt) acc[mi][nt] = vzero;

    for (int tap = 0; tap < 9; ++tap) {
        const int dy = tap / 3, dx = tap % 3;
        // B fragments: lane = output channel o, K = 32 input channels
        v16bf bfr[4];
        #pragma unroll
        for (int nt = 0; nt < 4; ++nt) {
            int o = nt * 16 + lane15;
            const uint4* wp = (const uint4*)(w2t + ((tap * 64 + o) * 32 + grp * 8));
            Frag32B fb;
            fb.u[0] = wp[0];
            fb.u[1] = wp[2];
            bfr[nt] = fb.v;
        }
        #pragma unroll
        for (int mi = 0; mi < 2; ++mi) {
            int m = wv * 2 + mi;            // M-tile == spatial row y
            const bf16* hp = &h1s[((m + dy) * 18 + (lane15 + dx)) * 32 + grp * 8];
            Frag32B fa;
            fa.u[0] = *(const uint4*)hp;
            fa.u[1] = *(const uint4*)(hp + 16);
            #pragma unroll
            for (int nt = 0; nt < 4; ++nt) {
                acc[mi][nt] = __builtin_amdgcn_wmma_f32_16x16x32_bf16(
                    false, fa.v, false, bfr[nt], (short)0, acc[mi][nt], false, false);
            }
        }
    }
    __syncthreads();   // xs region dead; reuse as prepool

    // ---- Phase 4a: bias + relu -> prepool LDS [pos(256)][ch(64)] bf16 ----
    #pragma unroll
    for (int mi = 0; mi < 2; ++mi) {
        int m = wv * 2 + mi;
        #pragma unroll
        for (int nt = 0; nt < 4; ++nt) {
            int ch = nt * 16 + lane15;
            float bz = b2[ch];
            #pragma unroll
            for (int r = 0; r < 8; ++r) {
                int pos = m * 16 + r + grp * 8;     // y = m, x = r + grp*8
                float v = fmaxf(acc[mi][nt][r] + bz, 0.f);
                prepool[pos * 64 + ch] = (bf16)v;
            }
        }
    }
    __syncthreads();

    // ---- Phase 4b: 2x2 maxpool, emit f[b][c*64 + py*8 + px] (torch .view order) ----
    bf16* fout = f + (size_t)b * 4096;
    #pragma unroll
    for (int i = 0; i < 16; ++i) {
        int q = i * 256 + t;
        int c = q >> 6, rem = q & 63, py = rem >> 3, pxo = rem & 7;
        int base = ((py * 2) * 16 + pxo * 2) * 64 + c;
        float v0 = (float)prepool[base];
        float v1 = (float)prepool[base + 64];
        float v2 = (float)prepool[base + 16 * 64];
        float v3 = (float)prepool[base + 16 * 64 + 64];
        fout[q] = (bf16)fmaxf(fmaxf(v0, v1), fmaxf(v2, v3));
    }
}

// ---------------------------------------------------------------------------
// GEMM f[4096,4096] x ct^T[4096,1024] + argmin of (c2[n] - 2*score).
// Workgroup = 32 feature rows (2 M-tiles); 8 waves x 8 N-tiles = 1024 centroids.
// 16 WMMAs per K-step per wave vs 20 fragment loads -> better L2 economy.
// ---------------------------------------------------------------------------
__global__ void __launch_bounds__(256)
voronoi_argmin_kernel(const bf16* __restrict__ f,   // [4096][4096]
                      const bf16* __restrict__ ct,  // [1024][4096]
                      const float* __restrict__ c2, // [1024]
                      int* __restrict__ out)        // [4096]
{
    const int t = threadIdx.x;
    const int wv = t >> 5, lane = t & 31;
    const int lane15 = lane & 15, grp = lane >> 4;
    const int r0 = blockIdx.x * 32;

    v8f acc[2][8];
    v8f vzero = {0.f, 0.f, 0.f, 0.f, 0.f, 0.f, 0.f, 0.f};
    #pragma unroll
    for (int mt = 0; mt < 2; ++mt)
        #pragma unroll
        for (int j = 0; j < 8; ++j) acc[mt][j] = vzero;

    const bf16* fp0 = f + (size_t)(r0 + lane15) * 4096 + grp * 8;
    const bf16* fp1 = f + (size_t)(r0 + 16 + lane15) * 4096 + grp * 8;

    for (int kc = 0; kc < 128; ++kc) {
        const int k0 = kc * 32;
        // Speculative prefetch of the streamed feature rows (~8 K-steps ahead);
        // lowers to global_prefetch_b8, OOB translation failures dropped silently.
        __builtin_prefetch((const void*)(fp0 + k0 + 256), 0, 0);
        __builtin_prefetch((const void*)(fp1 + k0 + 256), 0, 0);

        Frag32B fa0, fa1;
        fa0.u[0] = *(const uint4*)(fp0 + k0);
        fa0.u[1] = *(const uint4*)(fp0 + k0 + 16);
        fa1.u[0] = *(const uint4*)(fp1 + k0);
        fa1.u[1] = *(const uint4*)(fp1 + k0 + 16);
        #pragma unroll
        for (int j = 0; j < 8; ++j) {
            int n = (wv * 8 + j) * 16 + lane15;
            const bf16* bp = ct + (size_t)n * 4096 + k0 + grp * 8;
            Frag32B fb;
            fb.u[0] = *(const uint4*)bp;
            fb.u[1] = *(const uint4*)(bp + 16);
            acc[0][j] = __builtin_amdgcn_wmma_f32_16x16x32_bf16(
                false, fa0.v, false, fb.v, (short)0, acc[0][j], false, false);
            acc[1][j] = __builtin_amdgcn_wmma_f32_16x16x32_bf16(
                false, fa1.v, false, fb.v, (short)0, acc[1][j], false, false);
        }
    }

    // Lexicographic (value, index) argmin -> matches first-occurrence tie-break.
    __shared__ float bestv[32][128];
    __shared__ int   besti[32][128];

    float c2v[8]; int nidx[8];
    #pragma unroll
    for (int j = 0; j < 8; ++j) {
        nidx[j] = (wv * 8 + j) * 16 + lane15;
        c2v[j]  = c2[nidx[j]];
    }
    #pragma unroll
    for (int mt = 0; mt < 2; ++mt) {
        #pragma unroll
        for (int rr = 0; rr < 8; ++rr) {
            float bv = BIGF; int bi = 0x7fffffff;
            #pragma unroll
            for (int j = 0; j < 8; ++j) {
                float v = c2v[j] - 2.0f * acc[mt][j][rr];
                if (v < bv || (v == bv && nidx[j] < bi)) { bv = v; bi = nidx[j]; }
            }
            int row = mt * 16 + rr + grp * 8;
            bestv[row][wv * 16 + lane15] = bv;
            besti[row][wv * 16 + lane15] = bi;
        }
    }
    __syncthreads();

    if (t < 32) {
        float bv = BIGF; int bi = 0x7fffffff;
        for (int c = 0; c < 128; ++c) {
            float v = bestv[t][c]; int n = besti[t][c];
            if (v < bv || (v == bv && n < bi)) { bv = v; bi = n; }
        }
        out[r0 + t] = bi;
    }
}

// ---------------------------------------------------------------------------
extern "C" void kernel_launch(void* const* d_in, const int* in_sizes, int n_in,
                              void* d_out, int out_size, void* d_ws, size_t ws_size,
                              hipStream_t stream) {
    const float* x   = (const float*)d_in[0];  // [4096,3,32,32]
    const float* w1  = (const float*)d_in[1];  // [32,3,3,3]
    const float* b1  = (const float*)d_in[2];  // [32]
    const float* w2  = (const float*)d_in[3];  // [64,32,3,3]
    const float* b2  = (const float*)d_in[4];  // [64]
    const float* cen = (const float*)d_in[5];  // [1000,4096]

    char* ws = (char*)d_ws;
    bf16*  f_feat = (bf16*)ws;                                   // 32 MB
    bf16*  ct     = (bf16*)(ws + 33554432);                      // 8 MB
    bf16*  w2t    = (bf16*)(ws + 33554432 + 8388608);            // 36 KB
    float* c2     = (float*)(ws + 33554432 + 8388608 + 65536);   // 4 KB

    prep_w2_kernel<<<72, 256, 0, stream>>>(w2, w2t);
    prep_cent_kernel<<<1024, 256, 0, stream>>>(cen, ct, c2);
    conv_fused_kernel<<<4096, 256, 0, stream>>>(x, w1, b1, b2, w2t, f_feat);
    voronoi_argmin_kernel<<<128, 256, 0, stream>>>(f_feat, ct, c2, (int*)d_out);
}
